// NeuromorphicSRWKVJax_83734682403275
// MI455X (gfx1250) — matrix operand order repaired and checked by reading, not compiled, not run on hardware
//
#include <hip/hip_runtime.h>
#include <stdint.h>

// ---------- CDNA5 WMMA types ----------
typedef __bf16 v16bf __attribute__((ext_vector_type(16)));
typedef float  v8f   __attribute__((ext_vector_type(8)));

#define WMMA_BF16(A, B, C) \
  __builtin_amdgcn_wmma_f32_16x16x32_bf16(false, (A), false, (B), (short)0, (C), false, false)

static __device__ __forceinline__ uint16_t f2bf_bits(float f) {
  uint32_t u = __builtin_bit_cast(uint32_t, f);
  u += 0x7fffu + ((u >> 16) & 1u);          // round-to-nearest-even
  return (uint16_t)(u >> 16);
}
static __device__ __forceinline__ uint32_t pack2(float a, float b) {
  return (uint32_t)f2bf_bits(a) | ((uint32_t)f2bf_bits(b) << 16);
}

// A WMMA 16-bit operand is exactly 32 contiguous bytes of register file.
// Build it from two 16-byte loads with zero VALU work.
struct U32x8 { uint4 lo; uint4 hi; };
static __device__ __forceinline__ v16bf frag_from(uint4 lo, uint4 hi) {
  U32x8 t{lo, hi};
  return __builtin_bit_cast(v16bf, t);
}

// Problem constants: B=4, T=8192, D=512, H=16, Dh=32, BLOCK_KV=64.

// =====================================================================
// Kernel 0: one-shot fp32 -> bf16 down-convert of x (memory bound).
// =====================================================================
__global__ __launch_bounds__(256) void k_cvt_bf16(const float* __restrict__ x,
                                                  uint16_t* __restrict__ xb) {
  const int i = blockIdx.x * 256 + threadIdx.x;     // grid sized exactly
  const float4 v = ((const float4*)x)[i];
  uint2 o;
  o.x = pack2(v.x, v.y);
  o.y = pack2(v.z, v.w);
  ((uint2*)xb)[i] = o;
}

// =====================================================================
// Kernel 1: fused QKV projection.  xb[32768,512]bf16 @ W[512,512]fp32 ->
// q,k bf16 natural layout; v written transposed vT[b,h,dh,t] bf16.
// grid (256, 24): y/8 selects {q,k,v}, y%8 selects 64-col strip.
// =====================================================================
__global__ __launch_bounds__(256) void k_qkv_gemm(
    const uint16_t* __restrict__ xb, const float* __restrict__ Wq,
    const float* __restrict__ Wk, const float* __restrict__ Wv,
    uint16_t* __restrict__ qb, uint16_t* __restrict__ kb,
    uint16_t* __restrict__ vT) {
  __shared__ uint16_t WT[64][264];
  const int tid = threadIdx.x;
  const int lane = tid & 31, wv = tid >> 5;
  const int g = lane >> 4, ln = lane & 15;
  const int which = blockIdx.y >> 3;                 // 0=q 1=k 2=v
  const int ncb = (blockIdx.y & 7) << 6;             // column base
  const float* __restrict__ W = (which == 0) ? Wq : ((which == 1) ? Wk : Wv);
  const int rowbase = blockIdx.x * 128 + wv * 16;

  const v8f vz = {0.f, 0.f, 0.f, 0.f, 0.f, 0.f, 0.f, 0.f};
  v8f acc[4] = {vz, vz, vz, vz};

  for (int ph = 0; ph < 2; ++ph) {
    const int kbase = ph << 8;                       // split-K staging, 256 at a time
    for (int idx = tid; idx < 64 * 128; idx += 256) {
      const int n = idx & 63, kk = (idx >> 6) << 1;  // pack two k-rows per dword
      const uint32_t w = pack2(W[(size_t)(kbase + kk) * 512 + ncb + n],
                               W[(size_t)(kbase + kk + 1) * 512 + ncb + n]);
      *(uint32_t*)&WT[n][kk] = w;
    }
    __syncthreads();
    const int arow = rowbase + ln;
#pragma unroll
    for (int ks = 0; ks < 8; ++ks) {
      const int k0 = kbase + ks * 32;
      // A fragment: two 16B runs in contiguous k order (k = 16*g + s, as B)
      const uint4* a0 = (const uint4*)(xb + (size_t)arow * 512 + k0 + g * 16);
      const uint4* a1 = (const uint4*)(xb + (size_t)arow * 512 + k0 + 8 + g * 16);
      const v16bf af = frag_from(a0[0], a1[0]);
      // Preload all 4 B fragments; the sched barrier keeps the ds_load
      // clause ahead of the WMMA chain so waits retire progressively.
      v16bf bfr[4];
#pragma unroll
      for (int nt = 0; nt < 4; ++nt) {
        const uint4* wp = (const uint4*)&WT[nt * 16 + ln][ks * 32 + g * 16];
        bfr[nt] = frag_from(wp[0], wp[1]);
      }
      __builtin_amdgcn_sched_barrier(0);
#pragma unroll
      for (int nt = 0; nt < 4; ++nt) acc[nt] = WMMA_BF16(af, bfr[nt], acc[nt]);
    }
    __syncthreads();
  }
#pragma unroll
  for (int nt = 0; nt < 4; ++nt) {
#pragma unroll
    for (int r = 0; r < 8; ++r) {                    // C layout: row = r + 8*g
      const int col = ncb + nt * 16 + ln;
      const int row = rowbase + r + g * 8;
      const uint16_t hv = f2bf_bits(acc[nt][r]);
      if (which == 0) {
        qb[(size_t)row * 512 + col] = hv;
      } else if (which == 1) {
        kb[(size_t)row * 512 + col] = hv;
      } else {
        const int bb = row >> 13, t = row & 8191;
        const int h = col >> 5, dh = col & 31;
        vT[(size_t)((bb * 16 + h) * 32 + dh) * 8192 + t] = hv;
      }
    }
  }
}

// =====================================================================
// Kernel 2: flash attention.  grid (T/64=128, H=16, B=4), 128 threads
// (4 waves); each wave owns 16 q-rows.  Dh=32 == WMMA K -> one WMMA per
// 16x16 S tile.  Online softmax in exp2 domain (native v_exp_f32);
// P re-laid out C->A through a wave-private LDS slice (in-order DS).
// =====================================================================
__global__ __launch_bounds__(128) void k_flash_attn(
    const uint16_t* __restrict__ qb, const uint16_t* __restrict__ kb,
    const uint16_t* __restrict__ vT, uint16_t* __restrict__ am) {
  __shared__ uint16_t pbuf[4][16 * 64];
  const int tid = threadIdx.x, lane = tid & 31, wv = tid >> 5;
  const int g = lane >> 4, ln = lane & 15;
  const int b = blockIdx.z, h = blockIdx.y;
  const int qrow0 = blockIdx.x * 64 + wv * 16;
  const size_t qoff = ((size_t)b * 8192 + qrow0) * 512 + h * 32;

  // Q A-fragment (16x32 bf16), loaded once
  const uint4* q0 = (const uint4*)(qb + qoff + (size_t)ln * 512 + g * 16);
  const uint4* q1 = (const uint4*)(qb + qoff + (size_t)ln * 512 + 8 + g * 16);
  const v16bf qf = frag_from(q0[0], q1[0]);

  float m_row[8], z_row[8], alpha[8];
#pragma unroll
  for (int r = 0; r < 8; ++r) { m_row[r] = -__builtin_inff(); z_row[r] = 0.f; }
  const v8f vz = {0.f, 0.f, 0.f, 0.f, 0.f, 0.f, 0.f, 0.f};
  v8f o0 = vz, o1 = vz;
  // softmax in exp2 domain: scores scaled by (1/sqrt(32))*log2(e)
  const float scale2 = 0.17677669529663687f * 1.4426950408889634f;

  for (int kvb = 0; kvb < 128; ++kvb) {
    const int kv0 = kvb << 6;
    if (kvb + 1 < 128)                               // pull next K tile toward L2/WGP$
      __builtin_prefetch(kb + ((size_t)b * 8192 + kv0 + 64 + lane) * 512 + h * 32, 0, 0);

    // ---- S = Q K^T : preload 4 K B-fragments, then 4 WMMAs ----
    v16bf kf[4];
#pragma unroll
    for (int t = 0; t < 4; ++t) {
      const uint4* kp = (const uint4*)(kb + ((size_t)b * 8192 + kv0 + t * 16 + ln) * 512 + h * 32 + g * 16);
      kf[t] = frag_from(kp[0], kp[1]);
    }
    __builtin_amdgcn_sched_barrier(0);
    v8f s[4];
#pragma unroll
    for (int t = 0; t < 4; ++t) s[t] = WMMA_BF16(qf, kf[t], vz);
#pragma unroll
    for (int t = 0; t < 4; ++t) s[t] *= scale2;

    // ---- online softmax (row = r + 8*g per C layout), exp2 domain ----
#pragma unroll
    for (int r = 0; r < 8; ++r) {
      float mx = fmaxf(fmaxf(s[0][r], s[1][r]), fmaxf(s[2][r], s[3][r]));
      mx = fmaxf(mx, __shfl_xor(mx, 1, 32));
      mx = fmaxf(mx, __shfl_xor(mx, 2, 32));
      mx = fmaxf(mx, __shfl_xor(mx, 4, 32));
      mx = fmaxf(mx, __shfl_xor(mx, 8, 32));
      const float mn = fmaxf(m_row[r], mx);
      const float al = __builtin_amdgcn_exp2f(m_row[r] - mn);
      float zs = 0.f;
      const int prow = r + g * 8;
#pragma unroll
      for (int t = 0; t < 4; ++t) {
        const float p = __builtin_amdgcn_exp2f(s[t][r] - mn);
        zs += p;
        pbuf[wv][prow * 64 + t * 16 + ln] = f2bf_bits(p);
      }
      zs += __shfl_xor(zs, 1, 32);
      zs += __shfl_xor(zs, 2, 32);
      zs += __shfl_xor(zs, 4, 32);
      zs += __shfl_xor(zs, 8, 32);
      z_row[r] = z_row[r] * al + zs;
      m_row[r] = mn;
      alpha[r] = al;
    }
    // pbuf slice is wave-private and DS ops are in-order within a wave:
    // a compiler-level barrier is sufficient (no s_barrier needed).
    asm volatile("" ::: "memory");

#pragma unroll
    for (int r = 0; r < 8; ++r) { o0[r] *= alpha[r]; o1[r] *= alpha[r]; }

    // ---- O += P V : preload P/V fragments, then 4 WMMAs ----
    v16bf pf[2], vf[4];
#pragma unroll
    for (int c = 0; c < 2; ++c) {
      const uint4 plo = *(const uint4*)&pbuf[wv][ln * 64 + c * 32 + g * 16];
      const uint4 phi = *(const uint4*)&pbuf[wv][ln * 64 + c * 32 + 8 + g * 16];
      pf[c] = frag_from(plo, phi);
#pragma unroll
      for (int dt = 0; dt < 2; ++dt) {
        const uint4* vp = (const uint4*)(vT + (size_t)((b * 16 + h) * 32 + dt * 16 + ln) * 8192 + kv0 + c * 32 + g * 16);
        vf[c * 2 + dt] = frag_from(vp[0], vp[1]);
      }
    }
    __builtin_amdgcn_sched_barrier(0);
    o0 = WMMA_BF16(pf[0], vf[0], o0);
    o1 = WMMA_BF16(pf[0], vf[1], o1);
    o0 = WMMA_BF16(pf[1], vf[2], o0);
    o1 = WMMA_BF16(pf[1], vf[3], o1);
    asm volatile("" ::: "memory");
  }

  // ---- normalize and write merged-heads bf16 ----
#pragma unroll
  for (int r = 0; r < 8; ++r) {
    const float inv = 1.0f / (z_row[r] + 1e-9f);
    const int row = qrow0 + r + g * 8;
    const size_t ob = ((size_t)b * 8192 + row) * 512 + h * 32;
    am[ob + ln]      = f2bf_bits(o0[r] * inv);
    am[ob + 16 + ln] = f2bf_bits(o1[r] * inv);
  }
}

// =====================================================================
// Kernel 3: output projection.  am[32768,512]bf16 @ Wo[512,512]fp32 + bo
// -> out fp32.  Same structure as kernel 1.
// =====================================================================
__global__ __launch_bounds__(256) void k_out_gemm(
    const uint16_t* __restrict__ am, const float* __restrict__ Wo,
    const float* __restrict__ bo, float* __restrict__ out) {
  __shared__ uint16_t WT[64][264];
  const int tid = threadIdx.x, lane = tid & 31, wv = tid >> 5;
  const int g = lane >> 4, ln = lane & 15;
  const int ncb = blockIdx.y << 6;
  const int rowbase = blockIdx.x * 128 + wv * 16;

  const v8f vz = {0.f, 0.f, 0.f, 0.f, 0.f, 0.f, 0.f, 0.f};
  v8f acc[4] = {vz, vz, vz, vz};

  for (int ph = 0; ph < 2; ++ph) {
    const int kbase = ph << 8;
    for (int idx = tid; idx < 64 * 128; idx += 256) {
      const int n = idx & 63, kk = (idx >> 6) << 1;
      const uint32_t w = pack2(Wo[(size_t)(kbase + kk) * 512 + ncb + n],
                               Wo[(size_t)(kbase + kk + 1) * 512 + ncb + n]);
      *(uint32_t*)&WT[n][kk] = w;
    }
    __syncthreads();
    const int arow = rowbase + ln;
#pragma unroll
    for (int ks = 0; ks < 8; ++ks) {
      const int k0 = kbase + ks * 32;
      const uint4* a0 = (const uint4*)(am + (size_t)arow * 512 + k0 + g * 16);
      const uint4* a1 = (const uint4*)(am + (size_t)arow * 512 + k0 + 8 + g * 16);
      const v16bf af = frag_from(a0[0], a1[0]);
      v16bf bfr[4];
#pragma unroll
      for (int nt = 0; nt < 4; ++nt) {
        const uint4* wp = (const uint4*)&WT[nt * 16 + ln][ks * 32 + g * 16];
        bfr[nt] = frag_from(wp[0], wp[1]);
      }
      __builtin_amdgcn_sched_barrier(0);
#pragma unroll
      for (int nt = 0; nt < 4; ++nt) acc[nt] = WMMA_BF16(af, bfr[nt], acc[nt]);
    }
    __syncthreads();
  }
#pragma unroll
  for (int nt = 0; nt < 4; ++nt) {
#pragma unroll
    for (int r = 0; r < 8; ++r) {
      const int col = ncb + nt * 16 + ln;
      const int row = rowbase + r + g * 8;
      out[(size_t)row * 512 + col] = acc[nt][r] + bo[col];
    }
  }
}

// =====================================================================
extern "C" void kernel_launch(void* const* d_in, const int* in_sizes, int n_in,
                              void* d_out, int out_size, void* d_ws, size_t ws_size,
                              hipStream_t stream) {
  (void)in_sizes; (void)n_in; (void)out_size; (void)ws_size;
  const float* x  = (const float*)d_in[0];
  const float* Wq = (const float*)d_in[1];
  const float* Wk = (const float*)d_in[2];
  const float* Wv = (const float*)d_in[3];
  const float* Wo = (const float*)d_in[4];
  const float* bo = (const float*)d_in[5];
  float* out = (float*)d_out;

  // workspace: q + k + vT + attn_out + xb, all bf16 [32768 x 512] = 160 MiB
  uint16_t* qb = (uint16_t*)d_ws;
  uint16_t* kb = qb + (size_t)32768 * 512;
  uint16_t* vT = kb + (size_t)32768 * 512;
  uint16_t* am = vT + (size_t)32768 * 512;
  uint16_t* xb = am + (size_t)32768 * 512;

  k_cvt_bf16  <<<dim3(16384),      256, 0, stream>>>(x, xb);     // 32768*512/4 lanes
  k_qkv_gemm  <<<dim3(256, 24),    256, 0, stream>>>(xb, Wq, Wk, Wv, qb, kb, vT);
  k_flash_attn<<<dim3(128, 16, 4), 128, 0, stream>>>(qb, kb, vT, am);
  k_out_gemm  <<<dim3(256, 8),     256, 0, stream>>>(am, Wo, bo, out);
}
